// EncoderBlock_18176301597429
// MI455X (gfx1250) — matrix operand order repaired and checked by reading, not compiled, not run on hardware
//
#include <hip/hip_runtime.h>

// ---------------------------------------------------------------------------
// Types / helpers
// ---------------------------------------------------------------------------
typedef __attribute__((ext_vector_type(16))) __bf16 v16bf;
typedef __attribute__((ext_vector_type(8)))  float  v8f;

__device__ __forceinline__ unsigned short f2bf(float f) {
    union { float f; unsigned u; } v; v.f = f;
    unsigned r = v.u + 0x7fffu + ((v.u >> 16) & 1u);
    return (unsigned short)(r >> 16);
}

__device__ __forceinline__ v8f v8zero() {
    v8f z;
#pragma unroll
    for (int i = 0; i < 8; ++i) z[i] = 0.f;
    return z;
}

// A-matrix (16x32 bf16) per-lane k index for half h (0..15)
__device__ __forceinline__ int kmapA(int h, int lane) {
    return ((h >> 3) << 4) + (((lane >> 4) & 1) << 3) + (h & 7);
}
// B-matrix (32x16 bf16) per-lane k index for half h (0..15)
__device__ __forceinline__ int kmapB(int h, int lane) {
    return ((h >> 3) << 4) + (((lane >> 4) & 1) << 3) + (h & 7);
}

// Load A fragment from LDS tile stored [m][k] with row stride `stride` halfs.
__device__ __forceinline__ v16bf load_fragA(const unsigned short* b, int stride,
                                            int row, int k0, int lane) {
    const __bf16* p = reinterpret_cast<const __bf16*>(b) + (size_t)row * stride + k0;
    v16bf f;
#pragma unroll
    for (int h = 0; h < 16; ++h) f[h] = p[kmapA(h, lane)];
    return f;
}
// Load B fragment from LDS tile stored TRANSPOSED [n][k] with row stride `stride` halfs.
__device__ __forceinline__ v16bf load_fragB(const unsigned short* b, int stride,
                                            int row, int k0, int lane) {
    const __bf16* p = reinterpret_cast<const __bf16*>(b) + (size_t)row * stride + k0;
    v16bf f;
#pragma unroll
    for (int h = 0; h < 16; ++h) f[h] = p[kmapB(h, lane)];
    return f;
}

#define BDIM 16
#define CDIM 512
#define LDIM 1024
#define BCL  (BDIM * CDIM * LDIM)

// ---------------------------------------------------------------------------
// 1) x + timing_signal   (B,C,L) fp32
// ---------------------------------------------------------------------------
__global__ __launch_bounds__(256) void addts_kernel(const float* __restrict__ x,
                                                    float* __restrict__ out) {
    int idx = blockIdx.x * 256 + threadIdx.x;
    int l = idx & (LDIM - 1);
    int c = (idx >> 10) & (CDIM - 1);
    const float log_inc = 9.210340371976184f / 255.f; // ln(10000)/(nt-1), nt=256
    float sig;
    if (c < 256) { float inv = __expf(-(float)c * log_inc);        sig = __sinf((float)l * inv); }
    else         { float inv = __expf(-(float)(c - 256) * log_inc); sig = __cosf((float)l * inv); }
    out[idx] = x[idx] + sig;
}

// ---------------------------------------------------------------------------
// 2) LayerNorm over C for each (b,l); input (B,C,L) fp32; output fp32 or bf16
// ---------------------------------------------------------------------------
__global__ __launch_bounds__(256) void ln_kernel(const float* __restrict__ in,
                                                 float* __restrict__ outf,
                                                 unsigned short* __restrict__ outb,
                                                 const float* __restrict__ g,
                                                 const float* __restrict__ bt,
                                                 int writebf) {
    int idx = blockIdx.x * 256 + threadIdx.x; // over B*L
    int bb = idx >> 10;
    int l  = idx & (LDIM - 1);
    const float* p = in + (size_t)bb * CDIM * LDIM + l;
    float s = 0.f, s2 = 0.f;
    for (int c = 0; c < CDIM; ++c) { float v = p[(size_t)c * LDIM]; s += v; s2 += v * v; }
    float mean = s * (1.f / CDIM);
    float var  = s2 * (1.f / CDIM) - mean * mean;
    float rstd = rsqrtf(var + 1e-5f);
    for (int c = 0; c < CDIM; ++c) {
        float v = (p[(size_t)c * LDIM] - mean) * rstd * g[c] + bt[c];
        size_t o = (size_t)bb * CDIM * LDIM + (size_t)c * LDIM + l;
        if (writebf) outb[o] = f2bf(v);
        else         outf[o] = v;
    }
}

// ---------------------------------------------------------------------------
// 3) Depthwise conv K=7 along L, output packed to bf16
// ---------------------------------------------------------------------------
__global__ __launch_bounds__(256) void dwconv_kernel(const float* __restrict__ in,
                                                     const float* __restrict__ w, // [C][7]
                                                     unsigned short* __restrict__ outb) {
    int idx = blockIdx.x * 256 + threadIdx.x; // over B*C*L
    int l  = idx & (LDIM - 1);
    int bc = idx >> 10;
    int c  = bc & (CDIM - 1);
    const float* p = in + ((size_t)bc << 10);
    float acc = 0.f;
#pragma unroll
    for (int k = 0; k < 7; ++k) {
        int ll = l + k - 3;
        if (ll >= 0 && ll < LDIM) acc += w[c * 7 + k] * p[ll];
    }
    outb[idx] = f2bf(acc);
}

// ---------------------------------------------------------------------------
// 4) fp32 -> bf16 convert (weights)
// ---------------------------------------------------------------------------
__global__ __launch_bounds__(256) void cvt_kernel(const float* __restrict__ in,
                                                  unsigned short* __restrict__ out, int n) {
    int i = blockIdx.x * 256 + threadIdx.x;
    if (i < n) out[i] = f2bf(in[i]);
}

// ---------------------------------------------------------------------------
// 5) WMMA GEMM: out[b,o,l] = act( (sum_c W[o,c] * X[b,c,l] + bias[o]) * scale )
//    W: bf16 (O x C) row-major.  X: bf16 (B,C,L).  Tiles 128x128x32, 8 waves.
//    Double-buffered LDS: global loads for tile k+1 issued before computing
//    tile k; L2 warmed two tiles ahead with global_prefetch.
// ---------------------------------------------------------------------------
#define GLDA 40 // LDS row stride in halfs (32 + 8 pad, keeps 16B alignment)
template <int RELU, int BF16OUT>
__global__ __launch_bounds__(256) void gemm_wmma(const unsigned short* __restrict__ Wb,
                                                 const unsigned short* __restrict__ Xb,
                                                 const float* __restrict__ bias,
                                                 float* __restrict__ outf,
                                                 unsigned short* __restrict__ outb,
                                                 int O, int Cd, float scale) {
    __shared__ unsigned short As[2][128 * GLDA]; // W tile [m][k]
    __shared__ unsigned short Bs[2][128 * GLDA]; // X tile transposed [n][k]
    int tid = threadIdx.x, lane = tid & 31, wave = tid >> 5;
    int wm = wave >> 2, wn = wave & 3;
    int n0 = blockIdx.x * 128;
    int o0 = blockIdx.y * 128;
    int bb = n0 >> 10;
    int l0 = n0 & (LDIM - 1);
    const unsigned short* Xbase = Xb + (size_t)bb * Cd * LDIM + l0;
    int cn = lane & 15;

    // per-thread staging coordinates (two 16B chunks each for A and B)
    int rowA = tid >> 2;            // 0..63  (second chunk: +64)
    int colA = (tid & 3) << 3;      // 0,8,16,24
    int krB  = tid >> 4;            // 0..15  (second chunk: +16)
    int jcB  = (tid & 15) << 3;     // 0..120

    v8f acc[4][2];
#pragma unroll
    for (int i = 0; i < 4; ++i)
#pragma unroll
        for (int j = 0; j < 2; ++j) acc[i][j] = v8zero();

    uint4 ra0, ra1, rb0, rb1;
    auto load_regs = [&](int k0) {
        ra0 = *reinterpret_cast<const uint4*>(Wb + (size_t)(o0 + rowA) * Cd + k0 + colA);
        ra1 = *reinterpret_cast<const uint4*>(Wb + (size_t)(o0 + rowA + 64) * Cd + k0 + colA);
        rb0 = *reinterpret_cast<const uint4*>(Xbase + (size_t)(k0 + krB) * LDIM + jcB);
        rb1 = *reinterpret_cast<const uint4*>(Xbase + (size_t)(k0 + krB + 16) * LDIM + jcB);
    };
    auto store_lds = [&](int b) {
        *reinterpret_cast<uint4*>(&As[b][rowA * GLDA + colA]) = ra0;
        *reinterpret_cast<uint4*>(&As[b][(rowA + 64) * GLDA + colA]) = ra1;
        const unsigned short* h0 = reinterpret_cast<const unsigned short*>(&rb0);
        const unsigned short* h1 = reinterpret_cast<const unsigned short*>(&rb1);
#pragma unroll
        for (int jj = 0; jj < 8; ++jj) {
            Bs[b][(jcB + jj) * GLDA + krB]      = h0[jj];
            Bs[b][(jcB + jj) * GLDA + krB + 16] = h1[jj];
        }
    };

    load_regs(0);
    store_lds(0);
    __syncthreads();
    int buf = 0;
    for (int k0 = 0; k0 < Cd; k0 += 32) {
        bool more = (k0 + 32) < Cd;
        if (more) {
            load_regs(k0 + 32); // global loads in flight during compute below
            if (k0 + 64 < Cd) {
                __builtin_prefetch(Wb + (size_t)(o0 + rowA) * Cd + k0 + 64 + colA, 0, 0);
                __builtin_prefetch(Xbase + (size_t)(k0 + 64 + krB) * LDIM + jcB, 0, 0);
            }
        }
        v16bf afr[4], bfr[2];
#pragma unroll
        for (int mi = 0; mi < 4; ++mi)
            afr[mi] = load_fragA(As[buf], GLDA, wm * 64 + mi * 16 + cn, 0, lane);
#pragma unroll
        for (int ni = 0; ni < 2; ++ni)
            bfr[ni] = load_fragB(Bs[buf], GLDA, wn * 32 + ni * 16 + cn, 0, lane);
#pragma unroll
        for (int mi = 0; mi < 4; ++mi)
#pragma unroll
            for (int ni = 0; ni < 2; ++ni)
                acc[mi][ni] = __builtin_amdgcn_wmma_f32_16x16x32_bf16(
                    false, afr[mi], false, bfr[ni], (short)0, acc[mi][ni], false, false);
        if (more) {
            __syncthreads();       // everyone done reading buf^1 (from iter k-1)
            store_lds(buf ^ 1);
            __syncthreads();
            buf ^= 1;
        }
    }

    int hs = (lane >> 4) & 1;
#pragma unroll
    for (int mi = 0; mi < 4; ++mi)
#pragma unroll
        for (int ni = 0; ni < 2; ++ni)
#pragma unroll
            for (int r = 0; r < 8; ++r) {
                int o = o0 + wm * 64 + mi * 16 + r + 8 * hs;
                int lcol = l0 + wn * 32 + ni * 16 + cn;
                float v = (acc[mi][ni][r] + bias[o]) * scale;
                if (RELU) v = fmaxf(v, 0.f);
                size_t oidx = (size_t)bb * O * LDIM + (size_t)o * LDIM + lcol;
                if (BF16OUT) outb[oidx] = f2bf(v);
                else         outf[oidx] = v;
            }
}

// ---------------------------------------------------------------------------
// 6) Flash attention: grid (L/128, NUM_HEAD, B), 256 threads (8 waves).
//    Q:(B,512,L) fp32 (pre-scaled), KV:(B,1024,L) fp32 (K rows 0..511, V 512..1023)
//    out: (B,512,L) fp32 channel-major.
// ---------------------------------------------------------------------------
#define QSTR 72   // 64 + 8 pad
#define VSTR 136  // 128 + 8 pad
__global__ __launch_bounds__(256) void attn_kernel(const float* __restrict__ qbuf,
                                                   const float* __restrict__ kvbuf,
                                                   const int* __restrict__ mask,
                                                   float* __restrict__ outp) {
    // Ks: 128*72, Vs: 64*136, QP (Q aliased with P): 128*136 halfs
    __shared__ unsigned short Ks[128 * QSTR];
    __shared__ unsigned short Vs[64 * VSTR];
    __shared__ unsigned short QP[128 * VSTR];
    __shared__ float Ms[128];

    const int L = LDIM;
    int tid = threadIdx.x, lane = tid & 31, wave = tid >> 5;
    int cn = lane & 15, hs = (lane >> 4) & 1;
    int qi0  = blockIdx.x * 128;
    int head = blockIdx.y;
    int bb   = blockIdx.z;
    const float* Qg = qbuf  + (size_t)bb * 512 * L + (size_t)head * 64 * L;
    const float* Kg = kvbuf + (size_t)bb * 1024 * L + (size_t)head * 64 * L;
    const float* Vg = kvbuf + (size_t)bb * 1024 * L + (size_t)(512 + head * 64) * L;

    // stage Q tile transposed: QP[i][d] (stride QSTR while used as Q)
    for (int e = tid; e < 64 * 128; e += 256) {
        int d = e >> 7, i = e & 127;
        QP[i * QSTR + d] = f2bf(Qg[(size_t)d * L + qi0 + i]);
    }
    __syncthreads();
    v16bf aQ[2];
#pragma unroll
    for (int ks = 0; ks < 2; ++ks) aQ[ks] = load_fragA(QP, QSTR, wave * 16 + cn, ks * 32, lane);

    float m_run[8], l_run[8], alpha[8];
    v8f oacc[4];
#pragma unroll
    for (int r = 0; r < 8; ++r) { m_run[r] = -3.0e38f; l_run[r] = 0.f; }
#pragma unroll
    for (int nf = 0; nf < 4; ++nf) oacc[nf] = v8zero();

    for (int kc = 0; kc < L; kc += 128) {
        __syncthreads(); // previous iteration done with Ks/Vs/Ms (and Q frags cached)
        for (int e = tid; e < 64 * 128; e += 256) {
            int d = e >> 7, j = e & 127;
            Ks[j * QSTR + d] = f2bf(Kg[(size_t)d * L + kc + j]);
            Vs[d * VSTR + j] = f2bf(Vg[(size_t)d * L + kc + j]);
        }
        if (tid < 128) Ms[tid] = (float)mask[bb * L + kc + tid];
        __syncthreads();

        // S strip: 16 rows x 128 cols per wave
        v8f sfr[8];
#pragma unroll
        for (int jt = 0; jt < 8; ++jt) {
            v8f s = v8zero();
#pragma unroll
            for (int ks = 0; ks < 2; ++ks) {
                v16bf bK = load_fragB(Ks, QSTR, jt * 16 + cn, ks * 32, lane);
                s = __builtin_amdgcn_wmma_f32_16x16x32_bf16(false, aQ[ks], false, bK,
                                                            (short)0, s, false, false);
            }
            float mj = Ms[jt * 16 + cn];
#pragma unroll
            for (int r = 0; r < 8; ++r) s[r] = s[r] * mj + (1.f - mj) * (-1e30f);
            sfr[jt] = s;
        }
        // online softmax: row max across 8 tiles + 16 lanes of this half
        float mx[8];
#pragma unroll
        for (int r = 0; r < 8; ++r) {
            float v = sfr[0][r];
#pragma unroll
            for (int jt = 1; jt < 8; ++jt) v = fmaxf(v, sfr[jt][r]);
            v = fmaxf(v, __shfl_xor(v, 1));
            v = fmaxf(v, __shfl_xor(v, 2));
            v = fmaxf(v, __shfl_xor(v, 4));
            v = fmaxf(v, __shfl_xor(v, 8));
            mx[r] = v;
        }
#pragma unroll
        for (int r = 0; r < 8; ++r) {
            float mn = fmaxf(m_run[r], mx[r]);
            alpha[r] = __expf(m_run[r] - mn);
            m_run[r] = mn;
        }
        // P = exp(S - m), accumulate row sums, stage P (wave-private region of QP)
        unsigned short* Ps = QP + wave * 16 * VSTR;
#pragma unroll
        for (int r = 0; r < 8; ++r) {
            float psum = 0.f;
#pragma unroll
            for (int jt = 0; jt < 8; ++jt) {
                float p = __expf(sfr[jt][r] - m_run[r]);
                psum += p;
                Ps[(r + 8 * hs) * VSTR + jt * 16 + cn] = f2bf(p);
            }
            l_run[r] = l_run[r] * alpha[r] + psum;
#pragma unroll
            for (int nf = 0; nf < 4; ++nf) oacc[nf][r] *= alpha[r];
        }
        // O += P x V   (K dim = 128 keys -> 4 wmma steps)
#pragma unroll
        for (int kt = 0; kt < 4; ++kt) {
            v16bf aP = load_fragA(Ps, VSTR, cn, kt * 32, lane);
#pragma unroll
            for (int nf = 0; nf < 4; ++nf) {
                v16bf bV = load_fragB(Vs, VSTR, nf * 16 + cn, kt * 32, lane);
                oacc[nf] = __builtin_amdgcn_wmma_f32_16x16x32_bf16(false, aP, false, bV,
                                                                   (short)0, oacc[nf], false, false);
            }
        }
    }
    // final: reduce row sums across the 16-lane half, normalize, write out
#pragma unroll
    for (int r = 0; r < 8; ++r) {
        float s = l_run[r];
        s += __shfl_xor(s, 1);
        s += __shfl_xor(s, 2);
        s += __shfl_xor(s, 4);
        s += __shfl_xor(s, 8);
        float inv = 1.f / s;
        int i = qi0 + wave * 16 + r + 8 * hs;
#pragma unroll
        for (int nf = 0; nf < 4; ++nf) {
            int d = nf * 16 + cn;
            outp[(size_t)bb * 512 * L + (size_t)(head * 64 + d) * L + i] = oacc[nf][r] * inv;
        }
    }
}

// ---------------------------------------------------------------------------
// Host-side orchestration
// ---------------------------------------------------------------------------
extern "C" void kernel_launch(void* const* d_in, const int* in_sizes, int n_in,
                              void* d_out, int out_size, void* d_ws, size_t ws_size,
                              hipStream_t stream) {
    const float* x    = (const float*)d_in[0];
    const int*   mask = (const int*)d_in[1];
    const float* dw_w = (const float*)d_in[4];
    const float* pw_w = (const float*)d_in[5];
    const float* pw_b = (const float*)d_in[6];
    const float* cg   = (const float*)d_in[7];
    const float* cb   = (const float*)d_in[8];
    const float* w_kv = (const float*)d_in[9];
    const float* b_kv = (const float*)d_in[10];
    const float* w_q  = (const float*)d_in[11];
    const float* b_q  = (const float*)d_in[12];
    const float* ag   = (const float*)d_in[13];
    const float* ab   = (const float*)d_in[14];
    const float* w1   = (const float*)d_in[15];
    const float* b1   = (const float*)d_in[16];
    const float* w2   = (const float*)d_in[17];
    const float* b2   = (const float*)d_in[18];
    const float* fg   = (const float*)d_in[19];
    const float* fb   = (const float*)d_in[20];

    // workspace carve-up (~161 MB total)
    char* ws = (char*)d_ws;
    float*          cur = (float*)(ws + 0);                       // 33.5 MB (B,C,L)
    float*          tmp = (float*)(ws + (size_t)33554432);        // 33.5 MB (also Q buffer)
    float*          qb  = tmp;
    float*          kvb = (float*)(ws + (size_t)67108864);        // 67 MB (B,2C,L)
    unsigned short* xb  = (unsigned short*)(ws + (size_t)134217728); // 16.8 MB bf16 acts
    unsigned short* xb2 = (unsigned short*)(ws + (size_t)150994944); // 16.8 MB bf16 acts
    unsigned short* wb  = (unsigned short*)(ws + (size_t)167772160); // 1 MB bf16 weights

    dim3 blk(256);
    dim3 gElem(BCL / 256);
    dim3 gLN((BDIM * LDIM) / 256);
    dim3 gGemm(128, 4);   // (B*L/128, O/128) for O=512
    dim3 gGemmKV(128, 8); // O=1024
    dim3 gAttn(LDIM / 128, 8, BDIM);

    // 1) add timing signal
    addts_kernel<<<gElem, blk, 0, stream>>>(x, cur);

    // 2) conv stack
    for (int i = 0; i < 4; ++i) {
        ln_kernel<<<gLN, blk, 0, stream>>>(cur, tmp, nullptr, cg + i * CDIM, cb + i * CDIM, 0);
        dwconv_kernel<<<gElem, blk, 0, stream>>>(tmp, dw_w + (size_t)i * CDIM * 7, xb);
        cvt_kernel<<<(CDIM * CDIM) / 256, blk, 0, stream>>>(pw_w + (size_t)i * CDIM * CDIM, wb, CDIM * CDIM);
        gemm_wmma<1, 0><<<gGemm, blk, 0, stream>>>(wb, xb, pw_b + i * CDIM, cur, nullptr,
                                                   CDIM, CDIM, 1.0f);
    }

    // 3) attention projections
    ln_kernel<<<gLN, blk, 0, stream>>>(cur, nullptr, xb, ag, ab, 1);
    cvt_kernel<<<(2 * CDIM * CDIM) / 256, blk, 0, stream>>>(w_kv, wb, 2 * CDIM * CDIM);
    gemm_wmma<0, 0><<<gGemmKV, blk, 0, stream>>>(wb, xb, b_kv, kvb, nullptr, 2 * CDIM, CDIM, 1.0f);
    cvt_kernel<<<(CDIM * CDIM) / 256, blk, 0, stream>>>(w_q, wb, CDIM * CDIM);
    gemm_wmma<0, 0><<<gGemm, blk, 0, stream>>>(wb, xb, b_q, qb, nullptr, CDIM, CDIM, 0.125f);

    // 4) flash attention -> cur
    attn_kernel<<<gAttn, blk, 0, stream>>>(qb, kvb, mask, cur);

    // 5) FFN
    ln_kernel<<<gLN, blk, 0, stream>>>(cur, nullptr, xb, fg, fb, 1);
    cvt_kernel<<<(CDIM * CDIM) / 256, blk, 0, stream>>>(w1, wb, CDIM * CDIM);
    gemm_wmma<1, 1><<<gGemm, blk, 0, stream>>>(wb, xb, b1, nullptr, xb2, CDIM, CDIM, 1.0f);
    cvt_kernel<<<(CDIM * CDIM) / 256, blk, 0, stream>>>(w2, wb, CDIM * CDIM);
    gemm_wmma<0, 0><<<gGemm, blk, 0, stream>>>(wb, xb2, b2, (float*)d_out, nullptr, CDIM, CDIM, 1.0f);
}
